// SpectralFrustumAdapter_21947282882662
// MI455X (gfx1250) — compile-verified, matrix operand order checked by reading, not run on hardware
//
#include <hip/hip_runtime.h>
#include <hip/hip_bf16.h>
#include <math.h>

#define NB    16     // batches
#define NPTS  8192   // points per batch
#define DDIM  256
#define BNCH  32     // bottleneck channels

typedef __bf16 bf16_t;
typedef bf16_t v16bf __attribute__((ext_vector_type(16)));
typedef float  v8f   __attribute__((ext_vector_type(8)));

// ---------------------------------------------------------------- morton sort
static __device__ __forceinline__ unsigned long long mpart(long long v) {
    unsigned long long u = (unsigned long long)v;
    u &= 2097151ULL;
    u |= (u << 32);
    u &= 8725724278095871ULL;
    u |= (u << 16) & 8725728556220671ULL;
    u |= (u << 8)  & 282506020581391ULL;
    u |= (u << 4)  & 294878547030211ULL;
    u |= (u << 2)  & 321685687669321ULL;
    return u;
}

__global__ __launch_bounds__(1024) void sfa_morton_sort(
        const float* __restrict__ coords, unsigned* __restrict__ order,
        float* __restrict__ rbuf) {
    __shared__ unsigned long long key[NPTS];          // 64 KB of 320 KB LDS
    const int b = blockIdx.x, tid = threadIdx.x;
    const float* cb = coords + (size_t)b * NPTS * 3;
    for (int i = tid; i < NPTS; i += 1024) {
        float cx = cb[i * 3 + 0], cy = cb[i * 3 + 1], cz = cb[i * 3 + 2];
        rbuf[(size_t)b * NPTS + i] = sqrtf(cx * cx + cy * cy + cz * cz);
        long long qx = (long long)(cx * 1023.0f);
        long long qy = (long long)(cy * 1023.0f);
        long long qz = (long long)(cz * 1023.0f);
        qx = qx < 0 ? 0 : (qx > 1023 ? 1023 : qx);
        qy = qy < 0 ? 0 : (qy > 1023 ? 1023 : qy);
        qz = qz < 0 ? 0 : (qz > 1023 ? 1023 : qz);
        unsigned long long code = mpart(qx) | (mpart(qy) << 1) | (mpart(qz) << 2);
        key[i] = (code << 13) | (unsigned long long)i;   // index in LSBs -> stable
    }
    __syncthreads();
    for (unsigned k = 2; k <= NPTS; k <<= 1) {
        for (unsigned j = k >> 1; j > 0; j >>= 1) {
            for (unsigned i = tid; i < NPTS; i += 1024) {
                unsigned ixj = i ^ j;
                if (ixj > i) {
                    unsigned long long a = key[i], c = key[ixj];
                    bool up = ((i & k) == 0);
                    if (up ? (a > c) : (a < c)) { key[i] = c; key[ixj] = a; }
                }
            }
            __syncthreads();
        }
    }
    for (int i = tid; i < NPTS; i += 1024)
        order[(size_t)b * NPTS + i] = (unsigned)(key[i] & 8191ULL);
}

// --------------------------------------------------- fold mix@Wu^T into Wcomb
__global__ void sfa_wcomb(const float* __restrict__ U, const float* __restrict__ V,
                          const float* __restrict__ Wu, float* __restrict__ Wcomb) {
    int e = blockIdx.x * 256 + threadIdx.x;          // 32 blocks * 256 = 8192
    int j = e >> 8, d = e & 255;
    float acc = 0.f;
    for (int k = 0; k < 32; ++k) {
        float m = 0.f;
        #pragma unroll
        for (int r = 0; r < 4; ++r) m += U[j * 4 + r] * V[k * 4 + r];
        acc += m * Wu[d * 32 + k];
    }
    Wcomb[j * 256 + d] = acc;
}

// ------------------------- GEMM1: z = relu([x_perm, r] @ Wd^T + bd) -> [B,BN,N]
__global__ __launch_bounds__(256) void sfa_gemm1(
        const float* __restrict__ x, const float* __restrict__ Wd,
        const float* __restrict__ bd, const unsigned* __restrict__ order,
        const float* __restrict__ rbuf, float* __restrict__ zT) {
    const int wave = threadIdx.x >> 5, lane = threadIdx.x & 31;
    const int lr = lane & 15;
    const bool hi = lane >= 16;
    const int tile = blockIdx.x * 8 + wave;
    const int rowStart = tile << 4;
    const int b  = rowStart >> 13;
    const int n0 = rowStart & (NPTS - 1);
    const int klo0 = hi ? 8 : 0, klo1 = hi ? 24 : 16;

    const unsigned ord = order[(size_t)b * NPTS + n0 + lr];
    const float4* rowp = (const float4*)(x + ((size_t)b * NPTS + ord) * DDIM);
    const float* w0 = Wd + (size_t)lr * 257;         // output channel j = lr
    const float* w1 = Wd + (size_t)(lr + 16) * 257;  // output channel j = lr+16

    v8f acc0 = {}, acc1 = {};
    #pragma unroll
    for (int c = 0; c < 8; ++c) {                    // K = 256 = 8 x 32
        float4 f0 = rowp[c * 8 + (klo0 >> 2)];
        float4 f1 = rowp[c * 8 + (klo0 >> 2) + 1];
        float4 f2 = rowp[c * 8 + (klo1 >> 2)];
        float4 f3 = rowp[c * 8 + (klo1 >> 2) + 1];
        v16bf a;
        a[0]=(bf16_t)f0.x;  a[1]=(bf16_t)f0.y;  a[2]=(bf16_t)f0.z;  a[3]=(bf16_t)f0.w;
        a[4]=(bf16_t)f1.x;  a[5]=(bf16_t)f1.y;  a[6]=(bf16_t)f1.z;  a[7]=(bf16_t)f1.w;
        a[8]=(bf16_t)f2.x;  a[9]=(bf16_t)f2.y;  a[10]=(bf16_t)f2.z; a[11]=(bf16_t)f2.w;
        a[12]=(bf16_t)f3.x; a[13]=(bf16_t)f3.y; a[14]=(bf16_t)f3.z; a[15]=(bf16_t)f3.w;
        v16bf b0, b1;
        const int kb = c * 32;
        #pragma unroll
        for (int t = 0; t < 8; ++t) {
            b0[t]     = (bf16_t)w0[kb + klo0 + t];
            b0[8 + t] = (bf16_t)w0[kb + klo1 + t];
            b1[t]     = (bf16_t)w1[kb + klo0 + t];
            b1[8 + t] = (bf16_t)w1[kb + klo1 + t];
        }
        acc0 = __builtin_amdgcn_wmma_f32_16x16x32_bf16(false, a, false, b0, (short)0, acc0, false, false);
        acc1 = __builtin_amdgcn_wmma_f32_16x16x32_bf16(false, a, false, b1, (short)0, acc1, false, false);
    }
    // exact f32 rank-1 update for the (unpermuted) radius column K=256, + bias + relu
    const float wd0 = w0[256], wd1 = w1[256];
    const float bd0 = bd[lr], bd1 = bd[lr + 16];
    const int mbase = n0 + (hi ? 8 : 0);
    float o0[8], o1[8];
    #pragma unroll
    for (int g = 0; g < 8; ++g) {
        float rv = rbuf[(size_t)b * NPTS + mbase + g];
        o0[g] = fmaxf(acc0[g] + rv * wd0 + bd0, 0.f);
        o1[g] = fmaxf(acc1[g] + rv * wd1 + bd1, 0.f);
    }
    float* z0 = zT + ((size_t)b * BNCH + lr)      * NPTS + mbase;
    float* z1 = zT + ((size_t)b * BNCH + lr + 16) * NPTS + mbase;
    float4 s;
    s.x=o0[0]; s.y=o0[1]; s.z=o0[2]; s.w=o0[3]; ((float4*)z0)[0] = s;
    s.x=o0[4]; s.y=o0[5]; s.z=o0[6]; s.w=o0[7]; ((float4*)z0)[1] = s;
    s.x=o1[0]; s.y=o1[1]; s.z=o1[2]; s.w=o1[3]; ((float4*)z1)[0] = s;
    s.x=o1[4]; s.y=o1[5]; s.z=o1[6]; s.w=o1[7]; ((float4*)z1)[1] = s;
}

// --------------------------------------------- spectral filter: in-LDS FFT 8192
static __device__ void fft8192(float* re, float* im, float sign, int tid) {
    for (int i = tid; i < NPTS; i += 256) {          // bit-reverse permutation
        unsigned rv = __brev((unsigned)i) >> 19;
        if (rv > (unsigned)i) {
            float t;
            t = re[i]; re[i] = re[rv]; re[rv] = t;
            t = im[i]; im[i] = im[rv]; im[rv] = t;
        }
    }
    __syncthreads();
    for (int len = 2; len <= NPTS; len <<= 1) {
        const int half = len >> 1;
        const float ang = sign * 6.2831853071795864769f / (float)len;
        for (int t = tid; t < NPTS / 2; t += 256) {
            const int pos = t & (half - 1);
            const int i0  = ((t >> (31 - __clz(half))) << (32 - __clz(half))) + pos; // blk*len+pos
            const int i1  = i0 + half;
            float s, c;
            __sincosf(ang * (float)pos, &s, &c);
            float xr = re[i1] * c - im[i1] * s;
            float xi = re[i1] * s + im[i1] * c;
            re[i1] = re[i0] - xr; im[i1] = im[i0] - xi;
            re[i0] += xr;         im[i0] += xi;
        }
        __syncthreads();
    }
}

__global__ __launch_bounds__(256) void sfa_fft_filter(
        float* __restrict__ zT, const float* __restrict__ logit_d) {
    __shared__ float re[NPTS];
    __shared__ float im[NPTS];
    const int b = blockIdx.x >> 5, k = blockIdx.x & 31, tid = threadIdx.x;
    float* row = zT + ((size_t)b * BNCH + k) * NPTS;
    for (int i = tid; i < NPTS; i += 256) { re[i] = row[i]; im[i] = 0.f; }
    __syncthreads();
    fft8192(re, im, -1.0f, tid);                     // forward fft
    const float d = fminf(fmaxf(logit_d[k], -10.f), 10.f);
    for (int i = tid; i < NPTS; i += 256) {          // H = exp(-w*d), w=2*pi*i/(N-1)
        float w = (6.2831853071795864769f * (float)i) / 8191.0f;
        float h = expf(-w * d);
        re[i] *= h; im[i] *= h;
    }
    __syncthreads();
    fft8192(re, im, 1.0f, tid);                      // inverse fft (unscaled)
    for (int i = tid; i < NPTS; i += 256)
        row[i] = re[i] * (1.0f / (float)NPTS);       // .real / N
}

// -------------------- GEMM2: out = x + y @ Wcomb + bu   (K=32 -> 1 WMMA/tile)
__global__ __launch_bounds__(256) void sfa_gemm2(
        const float* __restrict__ x, const float* __restrict__ yT,
        const float* __restrict__ Wcomb, const float* __restrict__ bu,
        float* __restrict__ out) {
    const int wave = threadIdx.x >> 5, lane = threadIdx.x & 31;
    const int lr = lane & 15;
    const bool hi = lane >= 16;
    const int tile = blockIdx.x * 8 + wave;
    const int rowStart = tile << 4;
    const int b  = rowStart >> 13;
    const int n0 = rowStart & (NPTS - 1);
    const int klo0 = hi ? 8 : 0, klo1 = hi ? 24 : 16;

    const float* yb = yT + (size_t)b * BNCH * NPTS + (n0 + lr);
    v16bf a;
    #pragma unroll
    for (int t = 0; t < 8; ++t) {
        a[t]     = (bf16_t)yb[(size_t)(klo0 + t) * NPTS];
        a[8 + t] = (bf16_t)yb[(size_t)(klo1 + t) * NPTS];
    }
    const int mbase = n0 + (hi ? 8 : 0);
    for (int dt = 0; dt < 16; ++dt) {
        const int d = dt * 16 + lr;
        v16bf bf;
        #pragma unroll
        for (int t = 0; t < 8; ++t) {
            bf[t]     = (bf16_t)Wcomb[(klo0 + t) * DDIM + d];
            bf[8 + t] = (bf16_t)Wcomb[(klo1 + t) * DDIM + d];
        }
        v8f acc = {};
        acc = __builtin_amdgcn_wmma_f32_16x16x32_bf16(false, a, false, bf, (short)0, acc, false, false);
        const float bub = bu[d];
        #pragma unroll
        for (int g = 0; g < 8; ++g) {
            size_t idx = ((size_t)b * NPTS + mbase + g) * DDIM + d;
            out[idx] = x[idx] + bub + acc[g];
        }
    }
}

// -------------------------------------------------------------------- launcher
extern "C" void kernel_launch(void* const* d_in, const int* in_sizes, int n_in,
                              void* d_out, int out_size, void* d_ws, size_t ws_size,
                              hipStream_t stream) {
    const float* x       = (const float*)d_in[0];
    const float* coords  = (const float*)d_in[1];
    const float* Wd      = (const float*)d_in[2];
    const float* bd      = (const float*)d_in[3];
    const float* U       = (const float*)d_in[4];
    const float* V       = (const float*)d_in[5];
    const float* logit_d = (const float*)d_in[6];
    const float* Wu      = (const float*)d_in[7];
    const float* bu      = (const float*)d_in[8];
    float* out = (float*)d_out;

    char* ws = (char*)d_ws;
    unsigned* order = (unsigned*)ws;                          // 512 KB
    float* rbuf  = (float*)(ws + 524288);                     // 512 KB
    float* Wcomb = (float*)(ws + 1048576);                    // 32 KB
    float* zT    = (float*)(ws + 1048576 + 65536);            // 16 MB, 16B aligned

    sfa_morton_sort<<<NB, 1024, 0, stream>>>(coords, order, rbuf);
    sfa_wcomb<<<32, 256, 0, stream>>>(U, V, Wu, Wcomb);
    sfa_gemm1<<<1024, 256, 0, stream>>>(x, Wd, bd, order, rbuf, zT);
    sfa_fft_filter<<<NB * BNCH, 256, 0, stream>>>(zT, logit_d);
    sfa_gemm2<<<1024, 256, 0, stream>>>(x, zT, Wcomb, bu, out);
}
